// DeepLSTM_42331197669629
// MI455X (gfx1250) — compile-verified, matrix operand order checked by reading, not run on hardware
//
#include <hip/hip_runtime.h>
#include <cstdint>

// ---------------- problem constants ----------------
#define B_      2048
#define MAXLEN_ 1000
#define H_      512
#define T_      21
#define NC_     10
#define G_      2048   // 4*H
#define H2_     1024   // 2*H
#define KPAD0_  1024   // MAXLEN padded to multiple of 64
#define KSTEP_  64     // K staged per barrier (two 32-wide WMMA sub-steps)
#define MROWS_  (T_ * B_)  // 43008

typedef __attribute__((ext_vector_type(16))) __bf16 v16bf;
typedef __attribute__((ext_vector_type(8)))  __bf16 v8bf;
typedef __attribute__((ext_vector_type(8)))  float  v8f;
typedef __attribute__((ext_vector_type(4)))  unsigned int v4u;
typedef __attribute__((ext_vector_type(8)))  int v8i;
typedef __attribute__((ext_vector_type(4)))  int v4i;

#if defined(__gfx1250__) && __has_builtin(__builtin_amdgcn_tensor_load_to_lds)
#define USE_TDM 1
#endif

// ---------------- TDM 2D tile load: global -> LDS (bf16 elements) ----------------
#ifdef USE_TDM
__device__ __forceinline__ void tdm_load_2d(unsigned lds_addr, const void* gptr,
                                            unsigned tile_x, unsigned tile_y,
                                            unsigned stride_elems)
{
    unsigned long long ga = (unsigned long long)(uintptr_t)gptr;
    v4u g0;
    g0[0] = 1u;                                   // count=1, user descriptor, no gather
    g0[1] = lds_addr;                             // LDS byte address
    g0[2] = (unsigned)(ga & 0xFFFFFFFFull);       // global_addr[31:0]
    g0[3] = (unsigned)((ga >> 32) & 0x1FFFFFFull) | (2u << 30);  // addr[56:32] | type=2

    unsigned td0 = stride_elems;                  // tensor_dim0 (>= tile_x, no OOB)
    unsigned td1 = 0x7FFFFFFFu;                   // tensor_dim1 huge (no OOB)
    v8i g1;
    g1[0] = (int)(1u << 16);                      // wg_mask=0, data_size=1 (2 bytes)
    g1[1] = (int)((td0 & 0xFFFFu) << 16);         // atomic_addr=0 | tensor_dim0 lo16
    g1[2] = (int)(((td0 >> 16) & 0xFFFFu) | ((td1 & 0xFFFFu) << 16));
    g1[3] = (int)(((td1 >> 16) & 0xFFFFu) | ((tile_x & 0xFFFFu) << 16)); // tile_dim0
    g1[4] = (int)(tile_y & 0xFFFFu);              // tile_dim1 | tile_dim2=0 (2D)
    g1[5] = (int)stride_elems;                    // tensor_dim0_stride lo32
    g1[6] = 0;                                    // stride hi16 | dim1_stride lo16
    g1[7] = 0;
    v4i g2 = {0, 0, 0, 0};                        // group2 (unused, 2D tensor)
    v4i g3 = {0, 0, 0, 0};                        // group3 (unused, 2D tensor)
    v8i g4 = {0, 0, 0, 0, 0, 0, 0, 0};            // clang-23 6-arg form: extra group
    __builtin_amdgcn_tensor_load_to_lds(g0, g1, g2, g3, g4, 0);
}
#endif

// ---------------- WMMA helpers (16x16x32 bf16, fp32 acc) ----------------
__device__ __forceinline__ v8f wmma_bf16(v16bf a, v16bf b, v8f c) {
    return __builtin_amdgcn_wmma_f32_16x16x32_bf16(false, a, false, b, (short)0, c,
                                                   false, false);
}

// Load one WMMA operand fragment from a 32-value contiguous run in LDS.
// p points at run_start + kb (kb = lane<16 ? 0 : 8): two 16B chunks at +0, +16 elems.
union FragU { v16bf v; v8bf h[2]; };
__device__ __forceinline__ v16bf load_frag(const __bf16* p) {
    FragU u;
    u.h[0] = *(const v8bf*)(p);
    u.h[1] = *(const v8bf*)(p + 16);
    return u.v;
}

// ---------------- generic bf16 GEMM: C[M,2048] = A[M,K] * Bp[K,2048] + init ----------------
// A: row-major [M][lda].  Bp: fragment-native packed [K/32][2048][32].
// Block: 256 threads = 8 waves; tile 64(M) x 128(N); wave does 2x2 WMMA tiles;
// KSTEP_=64 staged per barrier -> 8 WMMAs per wave per stage.
// Cinit != null -> accumulate onto existing fp32 buffer (recurrent step, in place)
// Cinit == null -> init from biasA[n] + biasB[n]   (input projections)
__global__ void gemm_bf16_kernel(const __bf16* __restrict__ A, int lda,
                                 const __bf16* __restrict__ Bp, int Ktot,
                                 const float* Cinit,
                                 const float* __restrict__ biasA,
                                 const float* __restrict__ biasB,
                                 float* Cout)
{
    __shared__ __align__(16) __bf16 As[64 * KSTEP_];   // [row][64]
    __shared__ __align__(16) __bf16 Bs[2 * 128 * 32];  // [ks][col][32]
    const int tid = threadIdx.x, lane = tid & 31, wave = tid >> 5;
    const int wm = (wave & 1) * 32, wn = (wave >> 1) * 32;
    const int r0 = blockIdx.x * 64, nb = blockIdx.y * 128;
    const int lr = lane & 15, kb = (lane >> 4) << 3;   // fragment lane mapping

    v8f c[2][2];
#pragma unroll
    for (int mi = 0; mi < 2; mi++)
#pragma unroll
        for (int ni = 0; ni < 2; ni++) {
            int mbase = r0 + wm + 16 * mi + kb;
            int ncol  = nb + wn + 16 * ni + lr;
            if (Cinit) {
#pragma unroll
                for (int v = 0; v < 8; v++)
                    c[mi][ni][v] = Cinit[(size_t)(mbase + v) * G_ + ncol];
            } else {
                float bv = biasA[ncol] + biasB[ncol];
#pragma unroll
                for (int v = 0; v < 8; v++) c[mi][ni][v] = bv;
            }
        }

    for (int k0 = 0; k0 < Ktot; k0 += KSTEP_) {
        const __bf16* bsrc = Bp + ((size_t)(k0 >> 5) * G_ + nb) * 32;
        __syncthreads();
#ifdef USE_TDM
        if (wave == 0) {
            tdm_load_2d((unsigned)(uintptr_t)(void*)As,
                        (const void*)(A + (size_t)r0 * lda + k0),
                        (unsigned)KSTEP_, 64u, (unsigned)lda);
            tdm_load_2d((unsigned)(uintptr_t)(void*)Bs,
                        (const void*)bsrc, 4096u, 2u, (unsigned)(G_ * 32));
            __builtin_amdgcn_s_wait_tensorcnt(0);
        }
#else
        // A: 64x64 = 4096 elems, 512 x v8bf chunks
        for (int cw = tid; cw < 512; cw += 256) {
            int row = cw >> 3, kk8 = (cw & 7) << 3;
            *(v8bf*)(As + row * KSTEP_ + kk8) =
                *(const v8bf*)(A + (size_t)(r0 + row) * lda + k0 + kk8);
        }
        // B: 2 contiguous 4096-elem planes, 1024 x v8bf chunks
        for (int cw = tid; cw < 1024; cw += 256) {
            int plane = cw >> 9, off8 = (cw & 511) << 3;
            *(v8bf*)(Bs + plane * 4096 + off8) =
                *(const v8bf*)(bsrc + (size_t)plane * G_ * 32 + off8);
        }
#endif
        __syncthreads();
#pragma unroll
        for (int ks = 0; ks < 2; ks++) {
            v16bf a0 = load_frag(As + (wm + lr) * KSTEP_ + ks * 32 + kb);
            v16bf a1 = load_frag(As + (wm + 16 + lr) * KSTEP_ + ks * 32 + kb);
            v16bf b0 = load_frag(Bs + ((ks << 7) + wn + lr) * 32 + kb);
            v16bf b1 = load_frag(Bs + ((ks << 7) + wn + 16 + lr) * 32 + kb);
            c[0][0] = wmma_bf16(a0, b0, c[0][0]);
            c[0][1] = wmma_bf16(a0, b1, c[0][1]);
            c[1][0] = wmma_bf16(a1, b0, c[1][0]);
            c[1][1] = wmma_bf16(a1, b1, c[1][1]);
        }
    }

#pragma unroll
    for (int mi = 0; mi < 2; mi++)
#pragma unroll
        for (int ni = 0; ni < 2; ni++) {
            int mbase = r0 + wm + 16 * mi + kb;
            int ncol  = nb + wn + 16 * ni + lr;
#pragma unroll
            for (int v = 0; v < 8; v++)
                Cout[(size_t)(mbase + v) * G_ + ncol] = c[mi][ni][v];
        }
}

// ---------------- layer-0 input projection: one-hot(x) GEMM ----------------
// gx[t*B+b, g] = sum_m [x[b,m]==t] * wih0[g,m] + b_ih + b_hh
// A tile (0/1 in bf16) built in LDS from x; B tile via TDM from packed weights.
__global__ void gx0_onehot_kernel(const int* __restrict__ x,
                                  const __bf16* __restrict__ Bp,   // packed [KPAD0/32][2048][32]
                                  const float* __restrict__ biasA,
                                  const float* __restrict__ biasB,
                                  float* __restrict__ gx)
{
    __shared__ __align__(16) __bf16 As[64 * KSTEP_];
    __shared__ __align__(16) __bf16 Bs[2 * 128 * 32];
    const int tid = threadIdx.x, lane = tid & 31, wave = tid >> 5;
    const int wm = (wave & 1) * 32, wn = (wave >> 1) * 32;
    const int r0 = blockIdx.x * 64, nb = blockIdx.y * 128;
    const int lr = lane & 15, kb = (lane >> 4) << 3;
    const int t  = r0 >> 11;        // r0 / B_, tile never crosses t boundary
    const int b0 = r0 & (B_ - 1);

    v8f c[2][2];
#pragma unroll
    for (int mi = 0; mi < 2; mi++)
#pragma unroll
        for (int ni = 0; ni < 2; ni++) {
            int ncol = nb + wn + 16 * ni + lr;
            float bv = biasA[ncol] + biasB[ncol];
#pragma unroll
            for (int v = 0; v < 8; v++) c[mi][ni][v] = bv;
        }

    for (int k0 = 0; k0 < KPAD0_; k0 += KSTEP_) {
        const __bf16* bsrc = Bp + ((size_t)(k0 >> 5) * G_ + nb) * 32;
        __syncthreads();
        // build one-hot A tile (64 rows x 64 k) from x: 16 elems per thread
#pragma unroll
        for (int i = 0; i < 16; i++) {
            int e = tid * 16 + i;
            int row = e >> 6, kk = e & 63;
            int m = k0 + kk;
            float v = 0.f;
            if (m < MAXLEN_)
                v = (x[(size_t)(b0 + row) * MAXLEN_ + m] == t) ? 1.f : 0.f;
            As[e] = (__bf16)v;
        }
#ifdef USE_TDM
        if (wave == 0) {
            tdm_load_2d((unsigned)(uintptr_t)(void*)Bs,
                        (const void*)bsrc, 4096u, 2u, (unsigned)(G_ * 32));
            __builtin_amdgcn_s_wait_tensorcnt(0);
        }
#else
        for (int cw = tid; cw < 1024; cw += 256) {
            int plane = cw >> 9, off8 = (cw & 511) << 3;
            *(v8bf*)(Bs + plane * 4096 + off8) =
                *(const v8bf*)(bsrc + (size_t)plane * G_ * 32 + off8);
        }
#endif
        __syncthreads();
#pragma unroll
        for (int ks = 0; ks < 2; ks++) {
            v16bf a0 = load_frag(As + (wm + lr) * KSTEP_ + ks * 32 + kb);
            v16bf a1 = load_frag(As + (wm + 16 + lr) * KSTEP_ + ks * 32 + kb);
            v16bf b0 = load_frag(Bs + ((ks << 7) + wn + lr) * 32 + kb);
            v16bf b1 = load_frag(Bs + ((ks << 7) + wn + 16 + lr) * 32 + kb);
            c[0][0] = wmma_bf16(a0, b0, c[0][0]);
            c[0][1] = wmma_bf16(a0, b1, c[0][1]);
            c[1][0] = wmma_bf16(a1, b0, c[1][0]);
            c[1][1] = wmma_bf16(a1, b1, c[1][1]);
        }
    }

#pragma unroll
    for (int mi = 0; mi < 2; mi++)
#pragma unroll
        for (int ni = 0; ni < 2; ni++) {
            int mbase = r0 + wm + 16 * mi + kb;
            int ncol  = nb + wn + 16 * ni + lr;
#pragma unroll
            for (int v = 0; v < 8; v++)
                gx[(size_t)(mbase + v) * G_ + ncol] = c[mi][ni][v];
        }
}

// ---- weight pack: fp32 [D][N=2048][Ksrc] -> bf16 fragment-tiled [D][Kpad/32][2048][32] ----
__global__ void pack_weights_kernel(const float* __restrict__ src, __bf16* __restrict__ dst,
                                    int nrows, int ksrc, int kpad)
{
    int id = blockIdx.x * 256 + threadIdx.x;       // index into dst layout
    int klo  = id & 31;
    int tmp  = id >> 5;
    int n    = tmp & (2048 - 1);
    int tmp2 = tmp >> 11;
    int kblk = tmp2 & ((kpad >> 5) - 1);
    int d    = tmp2 / (kpad >> 5);
    int k    = (kblk << 5) + klo;
    float v = (k < ksrc) ? src[(size_t)d * nrows * ksrc + (size_t)n * ksrc + k] : 0.f;
    dst[id] = (__bf16)v;
}

__global__ void zero_state_kernel(__bf16* h, float* c)
{
    int id = blockIdx.x * 256 + threadIdx.x;   // B*H total
    h[id] = (__bf16)0.f;
    c[id] = 0.f;
}

// ---------------- LSTM gate nonlinearity + state update ----------------
__global__ void lstm_elem_kernel(const float* __restrict__ gates,  // [B][4H] (i,f,g,o)
                                 float* __restrict__ c, __bf16* __restrict__ h,
                                 __bf16* out_bf, float* out_f)     // row stride H2_
{
    int id = blockIdx.x * 256 + threadIdx.x;   // B*H total
    int b = id >> 9, j = id & (H_ - 1);
    const float* g = gates + (size_t)b * G_;
    float gi = g[j], gf = g[H_ + j], gg = g[2 * H_ + j], go = g[3 * H_ + j];
    float si = 1.f / (1.f + __expf(-gi));
    float sf = 1.f / (1.f + __expf(-gf));
    float so = 1.f / (1.f + __expf(-go));
    float cn = sf * c[id] + si * tanhf(gg);
    float hn = so * tanhf(cn);
    c[id] = cn;
    h[id] = (__bf16)hn;
    if (out_bf) out_bf[(size_t)b * H2_ + j] = (__bf16)hn;
    if (out_f)  out_f [(size_t)b * H2_ + j] = hn;
}

// ---------------- emb: z[b,k] = sum_t emb_w[t]*seq2[t,b,k] + emb_b ----------------
__global__ void emb_reduce_kernel(const float* __restrict__ seq2,
                                  const float* __restrict__ emb_w,
                                  const float* __restrict__ emb_b,
                                  float* __restrict__ z)
{
    int id = blockIdx.x * 256 + threadIdx.x;   // B*H2 total
    float acc = emb_b[0];
#pragma unroll
    for (int t = 0; t < T_; t++)
        acc += emb_w[t] * seq2[(size_t)t * B_ * H2_ + id];
    z[id] = acc;
}

// ---------------- fc: out[b,n] = z[b,:] . fc_w[n,:] + fc_b[n] ----------------
__global__ void fc_kernel(const float* __restrict__ z, const float* __restrict__ fc_w,
                          const float* __restrict__ fc_b, float* __restrict__ out)
{
    int id = blockIdx.x * 256 + threadIdx.x;
    if (id >= B_ * NC_) return;
    int b = id / NC_, n = id - b * NC_;
    const float* zr = z + (size_t)b * H2_;
    const float* wr = fc_w + (size_t)n * H2_;
    float acc = fc_b[n];
    for (int k = 0; k < H2_; k++) acc += zr[k] * wr[k];
    out[id] = acc;
}

// ---------------- workspace layout (bytes; all 256B aligned) ----------------
static constexpr size_t OFF_WIH0T = 0;                                        // 8 MB
static constexpr size_t OFF_WHH0T = OFF_WIH0T + 2ull * KPAD0_ * G_ * 2;       // 4 MB
static constexpr size_t OFF_WIH1T = OFF_WHH0T + 2ull * H_ * G_ * 2;           // 8 MB
static constexpr size_t OFF_WHH1T = OFF_WIH1T + 2ull * H2_ * G_ * 2;          // 4 MB
static constexpr size_t OFF_SEQ1  = OFF_WHH1T + 2ull * H_ * G_ * 2;           // bf16 [T][B][2H]
static constexpr size_t OFF_SEQ2  = OFF_SEQ1 + (size_t)T_ * B_ * H2_ * 2;     // f32  [T][B][2H]
static constexpr size_t OFF_GX    = OFF_SEQ2 + (size_t)T_ * B_ * H2_ * 4;     // f32  [T][B][4H]
static constexpr size_t OFF_H     = OFF_GX + (size_t)T_ * B_ * G_ * 4;        // bf16 [B][H]
static constexpr size_t OFF_C     = OFF_H + (size_t)B_ * H_ * 2;              // f32  [B][H]
static constexpr size_t OFF_Z     = OFF_C + (size_t)B_ * H_ * 4;              // f32  [B][2H]

extern "C" void kernel_launch(void* const* d_in, const int* in_sizes, int n_in,
                              void* d_out, int out_size, void* d_ws, size_t ws_size,
                              hipStream_t stream)
{
    (void)in_sizes; (void)n_in; (void)out_size; (void)ws_size;
    const int*   x     = (const int*)  d_in[0];
    const float* w_ih0 = (const float*)d_in[1];
    const float* w_hh0 = (const float*)d_in[2];
    const float* b_ih0 = (const float*)d_in[3];
    const float* b_hh0 = (const float*)d_in[4];
    const float* w_ih1 = (const float*)d_in[5];
    const float* w_hh1 = (const float*)d_in[6];
    const float* b_ih1 = (const float*)d_in[7];
    const float* b_hh1 = (const float*)d_in[8];
    const float* emb_w = (const float*)d_in[9];
    const float* emb_b = (const float*)d_in[10];
    const float* fc_w  = (const float*)d_in[11];
    const float* fc_b  = (const float*)d_in[12];
    float* out = (float*)d_out;
    char*  ws  = (char*)d_ws;

    __bf16* wih0t = (__bf16*)(ws + OFF_WIH0T);
    __bf16* whh0t = (__bf16*)(ws + OFF_WHH0T);
    __bf16* wih1t = (__bf16*)(ws + OFF_WIH1T);
    __bf16* whh1t = (__bf16*)(ws + OFF_WHH1T);
    __bf16* seq1  = (__bf16*)(ws + OFF_SEQ1);
    float*  seq2  = (float*) (ws + OFF_SEQ2);
    float*  gx    = (float*) (ws + OFF_GX);
    __bf16* hbuf  = (__bf16*)(ws + OFF_H);
    float*  cbuf  = (float*) (ws + OFF_C);
    float*  zbuf  = (float*) (ws + OFF_Z);

    // 1) pack weights to fragment-tiled bf16 [D][K/32][2048][32]
    pack_weights_kernel<<<2 * KPAD0_ * G_ / 256, 256, 0, stream>>>(w_ih0, wih0t, G_, MAXLEN_, KPAD0_);
    pack_weights_kernel<<<2 * H_    * G_ / 256, 256, 0, stream>>>(w_hh0, whh0t, G_, H_, H_);
    pack_weights_kernel<<<2 * H2_   * G_ / 256, 256, 0, stream>>>(w_ih1, wih1t, G_, H2_, H2_);
    pack_weights_kernel<<<2 * H_    * G_ / 256, 256, 0, stream>>>(w_hh1, whh1t, G_, H_, H_);

    const dim3 gemm_big(MROWS_ / 64, G_ / 128);   // 672 x 16
    const dim3 gemm_step(B_ / 64, G_ / 128);      // 32 x 16
    const int  ew_blocks = B_ * H_ / 256;         // 4096

    // 2) layer 0 (bidirectional)
    for (int d = 0; d < 2; d++) {
        gx0_onehot_kernel<<<gemm_big, 256, 0, stream>>>(
            x, wih0t + (size_t)d * KPAD0_ * G_, b_ih0 + d * G_, b_hh0 + d * G_, gx);
        zero_state_kernel<<<ew_blocks, 256, 0, stream>>>(hbuf, cbuf);
        for (int s = 0; s < T_; s++) {
            int t = (d == 0) ? s : (T_ - 1 - s);
            float* gxt = gx + (size_t)t * B_ * G_;
            gemm_bf16_kernel<<<gemm_step, 256, 0, stream>>>(
                hbuf, H_, whh0t + (size_t)d * H_ * G_, H_, gxt, nullptr, nullptr, gxt);
            lstm_elem_kernel<<<ew_blocks, 256, 0, stream>>>(
                gxt, cbuf, hbuf, seq1 + (size_t)t * B_ * H2_ + d * H_, nullptr);
        }
    }

    // 3) layer 1 (bidirectional), input = seq1 bf16 [T*B][2H]
    for (int d = 0; d < 2; d++) {
        gemm_bf16_kernel<<<gemm_big, 256, 0, stream>>>(
            seq1, H2_, wih1t + (size_t)d * H2_ * G_, H2_,
            nullptr, b_ih1 + d * G_, b_hh1 + d * G_, gx);
        zero_state_kernel<<<ew_blocks, 256, 0, stream>>>(hbuf, cbuf);
        for (int s = 0; s < T_; s++) {
            int t = (d == 0) ? s : (T_ - 1 - s);
            float* gxt = gx + (size_t)t * B_ * G_;
            gemm_bf16_kernel<<<gemm_step, 256, 0, stream>>>(
                hbuf, H_, whh1t + (size_t)d * H_ * G_, H_, gxt, nullptr, nullptr, gxt);
            lstm_elem_kernel<<<ew_blocks, 256, 0, stream>>>(
                gxt, cbuf, hbuf, nullptr, seq2 + (size_t)t * B_ * H2_ + d * H_);
        }
    }

    // 4) emb reduction over t, then fc
    emb_reduce_kernel<<<B_ * H2_ / 256, 256, 0, stream>>>(seq2, emb_w, emb_b, zbuf);
    fc_kernel<<<(B_ * NC_ + 255) / 256, 256, 0, stream>>>(zbuf, fc_w, fc_b, out);
}
